// GroupedQueryAttention_77764677861646
// MI455X (gfx1250) — matrix operand order repaired and checked by reading, not compiled, not run on hardware
//
#include <hip/hip_runtime.h>
#include <hip/hip_bf16.h>

// ---------------------------------------------------------------------------
// GQA forward for MI455X (gfx1250, wave32, WMMA, async global->LDS copies).
// Pipeline: f32->bf16 convert (+ weight transpose) -> Q/K/V projections
//           (bf16 WMMA, V written transposed) -> scores QK^T (fp32 out,
//           scaled) -> row softmax -> P@V (P converted to bf16 on the fly)
//           -> output projection.
// attn_weight (second tuple output) is materialized in d_out after `out`.
// GEMM: 128x64 block tile, 8 waves, each wave owns a 32x32 region
// (4 accumulators, 2 LDS b128 loads per WMMA), double-buffered LDS filled
// with global_load_async_to_lds_b128 (ASYNCcnt) while WMMAs consume.
// ---------------------------------------------------------------------------

typedef __attribute__((ext_vector_type(16))) __bf16 bf16x16;
typedef __attribute__((ext_vector_type(8)))  float  f32x8;

#define BTM 128  // block tile M
#define BTN 64   // block tile N
#define KT  64   // k tile

constexpr int T_   = 2048;  // sequence length
constexpr int NH   = 16;    // query heads
constexpr int GRP  = 4;     // H / KH
constexpr int BTOT = 4096;  // B*T

union FragAB { uint4 q[2]; bf16x16 v; };

// gfx1250 async global->LDS copy (ASYNCcnt-tracked, no VGPR staging).
__device__ __forceinline__ void async_b128(unsigned lds, const void* g) {
  asm volatile("global_load_async_to_lds_b128 %0, %1, off"
               :: "v"(lds), "v"((unsigned long long)(uintptr_t)g)
               : "memory");
}
__device__ __forceinline__ void wait_async() {
  asm volatile("s_wait_asynccnt 0x0" ::: "memory");
}
// Low 32 bits of a generic LDS pointer == LDS byte offset (aperture layout).
__device__ __forceinline__ unsigned lds_off(const void* p) {
  return (unsigned)(uintptr_t)p;
}

// ---------------------------------------------------------------------------
__global__ __launch_bounds__(256)
void cvt_f32_bf16(const float* __restrict__ in, __bf16* __restrict__ out, long n) {
  long i      = (long)blockIdx.x * blockDim.x + threadIdx.x;
  long stride = (long)gridDim.x * blockDim.x;
  for (; i < n; i += stride) out[i] = (__bf16)in[i];
}

// Transpose [K,N] f32 -> [N,K] bf16 (32x32 tiles through padded LDS).
__global__ __launch_bounds__(256)
void cvt_transpose(const float* __restrict__ in, __bf16* __restrict__ out,
                   int K, int N) {
  __shared__ float tile[32][33];
  const int tx = threadIdx.x, ty = threadIdx.y;      // 32 x 8
  const int k0 = blockIdx.y * 32, n0 = blockIdx.x * 32;
  #pragma unroll
  for (int i = 0; i < 32; i += 8)
    tile[ty + i][tx] = in[(long)(k0 + ty + i) * N + n0 + tx];
  __syncthreads();
  #pragma unroll
  for (int i = 0; i < 32; i += 8)
    out[(long)(n0 + ty + i) * K + k0 + tx] = (__bf16)tile[tx][ty + i];
}

// ---------------------------------------------------------------------------
// MODE 0: plain GEMM (projections), gridDim.z == 1
// MODE 1: scores  S[z] = Q_head @ K_head^T      (z = b*NH + h)
// MODE 2: context C[z] = P[z] @ V_head          (z = b*NH + h, B = V^T)
// B operand is ALWAYS row-major [N,K] (weights pre-transposed, Kh natural,
// V written transposed) -> staged with async b128 copies into Bs[n][k].
// A_F32   : A operand fp32 in global, converted to bf16 while staged in LDS
// OUT_BF16: write bf16 output (else fp32)
// HAS_BIAS: add bias[n]
// OUT_TRANS: store C transposed (C[n][m], ldc = column stride)
template<int MODE, int A_F32, int OUT_BF16, int HAS_BIAS, int OUT_TRANS>
__global__ __launch_bounds__(256)
void gemm_wmma(const void* __restrict__ Ap, const __bf16* __restrict__ Bp,
               const float* __restrict__ bias, void* __restrict__ Cp,
               int M, int N, int K, int lda, int ldb, int ldc, float scale)
{
  (void)M; (void)N;
  __shared__ __bf16 As[2][BTM * KT];   // [m][k]
  __shared__ __bf16 Bs[2][BTN * KT];   // [n][k]  (k-contiguous)

  const int tid  = threadIdx.x;
  const int wave = tid >> 5;
  const int lane = tid & 31;
  const int z    = blockIdx.z;

  long aOff = 0, bOff = 0, cOff = 0;
  if (MODE == 1) {
    const int b = z / NH, h = z % NH;
    aOff = (long)b * T_ * 2048 + (long)h * 128;            // Qh slice
    bOff = (long)b * T_ * 512 + (long)(h / GRP) * 128;     // Kh slice
    cOff = (long)z * T_ * T_;                              // probs
  }
  if (MODE == 2) {
    const int b = z / NH, h = z % NH;
    aOff = (long)z * T_ * T_;                              // probs
    bOff = (long)(h / GRP) * 128 * BTOT + (long)b * T_;    // V^T slice
    cOff = (long)b * T_ * 2048 + (long)h * 128;            // ctx slice
  }

  const int m0 = blockIdx.y * BTM;
  const int n0 = blockIdx.x * BTN;

  // wave -> 32x32 region: 4 waves down M, 2 across N; 2x2 accumulators
  const int wm = (wave & 3) * 32;
  const int wn = (wave >> 2) * 32;

  f32x8 acc[2][2] = {};

  // async staging addressing: 16B chunks, 8 per 64-half row
  const int cr = tid >> 3;         // 0..31 (stepped by 32)
  const int cc = (tid & 7) << 3;   // 0,8,..,56 halves (16B aligned)
  // f32 staging (MODE2 A): 16 floats per thread per pass
  const int lr4  = tid >> 2;       // 0..63 (stepped by 64)
  const int lc16 = (tid & 3) << 4; // 0,16,32,48

  auto stageA = [&](int k0, int buf) {
    if (A_F32) {
      #pragma unroll
      for (int i = 0; i < 2; ++i) {
        const int r = lr4 + i * 64;
        const float* Ag = (const float*)Ap + aOff + (long)(m0 + r) * lda + k0 + lc16;
        __bf16* dst = &As[buf][r * KT + lc16];
        #pragma unroll
        for (int j = 0; j < 16; j += 4) {
          float4 f = *(const float4*)(Ag + j);
          dst[j + 0] = (__bf16)f.x; dst[j + 1] = (__bf16)f.y;
          dst[j + 2] = (__bf16)f.z; dst[j + 3] = (__bf16)f.w;
        }
      }
    } else {
      #pragma unroll
      for (int i = 0; i < 4; ++i) {            // 128 rows x 64 halves
        const int r = cr + i * 32;
        const __bf16* g = (const __bf16*)Ap + aOff + (long)(m0 + r) * lda + k0 + cc;
        async_b128(lds_off(&As[buf][r * KT + cc]), g);
      }
    }
  };
  auto stageB = [&](int k0, int buf) {
    #pragma unroll
    for (int i = 0; i < 2; ++i) {              // 64 rows x 64 halves
      const int r = cr + i * 32;
      const __bf16* g = Bp + bOff + (long)(n0 + r) * ldb + k0 + cc;
      async_b128(lds_off(&Bs[buf][r * KT + cc]), g);
    }
  };

  // prologue: fill buffer 0
  stageA(0, 0);
  stageB(0, 0);
  wait_async();
  __syncthreads();

  int p = 0;
  for (int k0 = 0; k0 < K; k0 += KT) {
    // kick off next tile into the other buffer while computing this one
    if (k0 + KT < K) {
      stageA(k0 + KT, p ^ 1);
      stageB(k0 + KT, p ^ 1);
    }

    const __bf16* Asp = As[p];
    const __bf16* Bsp = Bs[p];

    #pragma unroll
    for (int ks = 0; ks < KT; ks += 32) {
      // A fragments (16x32 bf16): m = lane%16;
      // elems 0..7:  k = ks + (lane>=16?8:0) + e ; elems 8..15: +16
      FragAB fa[2];
      const int kA = ks + ((lane & 16) ? 8 : 0);
      #pragma unroll
      for (int im = 0; im < 2; ++im) {
        const int m = wm + im * 16 + (lane & 15);
        fa[im].q[0] = *(const uint4*)&Asp[m * KT + kA];
        fa[im].q[1] = *(const uint4*)&Asp[m * KT + kA + 16];
      }
      // B fragments (32x16 bf16): n = lane%16;
      // elems 0..7:  k = ks + (lane>=16?8:0) + e ; elems 8..15: +16
      FragAB fb[2];
      const int kB = ks + ((lane & 16) ? 8 : 0);
      #pragma unroll
      for (int in = 0; in < 2; ++in) {
        const int n = wn + in * 16 + (lane & 15);
        const uint4* pb = (const uint4*)&Bsp[n * KT + kB];
        fb[in].q[0] = pb[0]; fb[in].q[1] = pb[2];
      }
      #pragma unroll
      for (int im = 0; im < 2; ++im)
        #pragma unroll
        for (int in = 0; in < 2; ++in)
          acc[im][in] = __builtin_amdgcn_wmma_f32_16x16x32_bf16(
              false, fa[im].v, false, fb[in].v, (short)0, acc[im][in],
              false, false);
    }

    // next buffer's async copies must have landed before anyone reads it,
    // and all waves must be done reading buffer p before it is refilled.
    wait_async();
    __syncthreads();
    p ^= 1;
  }

  // ---- writeback: VGPR v -> m = v + (lane>=16?8:0), n = lane%16 ----
  const int nl = lane & 15;
  const int mB = ((lane & 16) ? 8 : 0);
  #pragma unroll
  for (int im = 0; im < 2; ++im) {
    #pragma unroll
    for (int in = 0; in < 2; ++in) {
      #pragma unroll
      for (int vv = 0; vv < 8; ++vv) {
        const long mg = m0 + wm + im * 16 + mB + vv;
        const int  n  = n0 + wn + in * 16 + nl;
        float val = acc[im][in][vv] * scale;
        if (HAS_BIAS) val += bias[n];
        const long idx = OUT_TRANS ? (cOff + (long)n * ldc + mg)
                                   : (cOff + mg * ldc + n);
        if (OUT_BF16) ((__bf16*)Cp)[idx] = (__bf16)val;
        else          ((float*)Cp)[idx] = val;
      }
    }
  }
}

// ---------------------------------------------------------------------------
// Row softmax over 2048 columns, one 256-thread block per row (8 waves).
__global__ __launch_bounds__(256)
void softmax_rows(float* __restrict__ base) {
  float* p = base + (long)blockIdx.x * T_;
  const int tid = threadIdx.x;
  const int wv = tid >> 5, ln = tid & 31;

  float vv[8];
  float m = -3.402823466e38f;
  #pragma unroll
  for (int i = 0; i < 8; ++i) { vv[i] = p[tid + (i << 8)]; m = fmaxf(m, vv[i]); }
  #pragma unroll
  for (int o = 16; o; o >>= 1) m = fmaxf(m, __shfl_xor(m, o, 32));

  __shared__ float red[8];
  __shared__ float bcast;
  if (ln == 0) red[wv] = m;
  __syncthreads();
  if (tid == 0) {
    float t = red[0];
    for (int i = 1; i < 8; ++i) t = fmaxf(t, red[i]);
    bcast = t;
  }
  __syncthreads();
  m = bcast;

  float s = 0.f;
  #pragma unroll
  for (int i = 0; i < 8; ++i) { vv[i] = __expf(vv[i] - m); s += vv[i]; }
  #pragma unroll
  for (int o = 16; o; o >>= 1) s += __shfl_xor(s, o, 32);
  __syncthreads();
  if (ln == 0) red[wv] = s;
  __syncthreads();
  if (tid == 0) {
    float t = 0.f;
    for (int i = 0; i < 8; ++i) t += red[i];
    bcast = t;
  }
  __syncthreads();
  const float inv = 1.0f / bcast;
  #pragma unroll
  for (int i = 0; i < 8; ++i) p[tid + (i << 8)] = vv[i] * inv;
}

// ---------------------------------------------------------------------------
extern "C" void kernel_launch(void* const* d_in, const int* in_sizes, int n_in,
                              void* d_out, int out_size, void* d_ws, size_t ws_size,
                              hipStream_t stream) {
  (void)in_sizes; (void)n_in; (void)out_size; (void)ws_size;

  const float* q  = (const float*)d_in[0];
  const float* k  = (const float*)d_in[1];
  const float* v  = (const float*)d_in[2];
  const float* Wq = (const float*)d_in[3];
  const float* bq = (const float*)d_in[4];
  const float* Wk = (const float*)d_in[5];
  const float* bk = (const float*)d_in[6];
  const float* Wv = (const float*)d_in[7];
  const float* bv = (const float*)d_in[8];
  const float* Wo = (const float*)d_in[9];
  const float* bo = (const float*)d_in[10];

  const long nQKV = 2L * 2048 * 2048;   // B*T*2048
  const long nWq  = 2048L * 2048;
  const long nWk  = 2048L * 512;
  const long nKh  = 2L * 2048 * 512;    // B*T*512

  char* w = (char*)d_ws;
  auto alloc = [&](long bytes) -> void* {
    void* p = (void*)w;
    w += (bytes + 255) & ~255L;
    return p;
  };
  __bf16* qb  = (__bf16*)alloc(nQKV * 2);
  __bf16* kb  = (__bf16*)alloc(nQKV * 2);
  __bf16* vb  = (__bf16*)alloc(nQKV * 2);
  __bf16* Wqt = (__bf16*)alloc(nWq * 2);   // [2048,2048] = Wq^T
  __bf16* Wkt = (__bf16*)alloc(nWk * 2);   // [512,2048]  = Wk^T
  __bf16* Wvt = (__bf16*)alloc(nWk * 2);   // [512,2048]  = Wv^T
  __bf16* Wot = (__bf16*)alloc(nWq * 2);   // [2048,2048] = Wo^T
  __bf16* Qh  = (__bf16*)alloc(nQKV * 2);  // [4096,2048]
  __bf16* Kh  = (__bf16*)alloc(nKh * 2);   // [4096,512]
  __bf16* Vt  = (__bf16*)alloc(nKh * 2);   // [512,4096]  = Vh^T
  __bf16* ctx = (__bf16*)alloc(nQKV * 2);  // [4096,2048]

  float* out   = (float*)d_out;
  float* probs = out + nQKV;               // attn_weight region [B,H,T,T]

  dim3 blk(256);
  dim3 tblk(32, 8);

  cvt_f32_bf16<<<1024, blk, 0, stream>>>(q, qb, nQKV);
  cvt_f32_bf16<<<1024, blk, 0, stream>>>(k, kb, nQKV);
  cvt_f32_bf16<<<1024, blk, 0, stream>>>(v, vb, nQKV);
  cvt_transpose<<<dim3(2048 / 32, 2048 / 32), tblk, 0, stream>>>(Wq, Wqt, 2048, 2048);
  cvt_transpose<<<dim3(512 / 32, 2048 / 32), tblk, 0, stream>>>(Wk, Wkt, 2048, 512);
  cvt_transpose<<<dim3(512 / 32, 2048 / 32), tblk, 0, stream>>>(Wv, Wvt, 2048, 512);
  cvt_transpose<<<dim3(2048 / 32, 2048 / 32), tblk, 0, stream>>>(Wo, Wot, 2048, 2048);

  // Q/K/V projections (B operand = pre-transposed weights, [N,K])
  gemm_wmma<0,0,1,1,0><<<dim3(2048 / BTN, 4096 / BTM, 1), blk, 0, stream>>>(
      qb, Wqt, bq, Qh, 4096, 2048, 2048, 2048, 2048, 2048, 1.0f);
  gemm_wmma<0,0,1,1,0><<<dim3(512 / BTN, 4096 / BTM, 1), blk, 0, stream>>>(
      kb, Wkt, bk, Kh, 4096, 512, 2048, 2048, 2048, 512, 1.0f);
  // V projection writes V^T directly ([512, 4096], ldc = 4096)
  gemm_wmma<0,0,1,1,1><<<dim3(512 / BTN, 4096 / BTM, 1), blk, 0, stream>>>(
      vb, Wvt, bv, Vt, 4096, 512, 2048, 2048, 2048, 4096, 1.0f);

  // scores: S = Q_head @ K_head^T * dk^-0.5 (fp32 into attn_weight region)
  const float scl = 0.08838834764831845f;  // 128^-0.5
  gemm_wmma<1,0,0,0,0><<<dim3(2048 / BTN, 2048 / BTM, 32), blk, 0, stream>>>(
      Qh, Kh, nullptr, probs, 2048, 2048, 128, 2048, 512, 2048, scl);

  // softmax over keys (in place, B*H*T rows)
  softmax_rows<<<2 * 16 * 2048, blk, 0, stream>>>(probs);

  // context: C = P @ V_head  (P fp32 -> bf16 on the fly; B = V^T, ldb = 4096)
  gemm_wmma<2,1,1,0,0><<<dim3(128 / BTN, 2048 / BTM, 32), blk, 0, stream>>>(
      probs, Vt, nullptr, ctx, 2048, 128, 2048, 2048, 4096, 2048, 1.0f);

  // output projection: out = ctx @ Wo + bo (fp32 out)
  gemm_wmma<0,0,0,1,0><<<dim3(2048 / BTN, 4096 / BTM, 1), blk, 0, stream>>>(
      ctx, Wot, bo, out, 4096, 2048, 2048, 2048, 2048, 2048, 1.0f);
}
